// LayerNormLSTMCell_37812892074298
// MI455X (gfx1250) — matrix operand order repaired and checked by reading, not compiled, not run on hardware
//
#include <hip/hip_runtime.h>

// ---------------------------------------------------------------------------
// Types for CDNA5 WMMA (gfx1250, wave32)
// ---------------------------------------------------------------------------
typedef __attribute__((ext_vector_type(16))) __bf16 v16bf;
typedef __attribute__((ext_vector_type(8)))  float  v8f;
typedef __attribute__((ext_vector_type(4)))  __bf16 bf16x4;
typedef __attribute__((ext_vector_type(4)))  int    v4i;

union FragU {
    uint4 u[2];
    v16bf v;
};

// CDNA5 async global->LDS copy path (ASYNCcnt). Guarded: falls back to the
// sync staging path if the toolchain lacks the builtins.
#ifndef __has_builtin
#define __has_builtin(x) 0
#endif
#if __has_builtin(__builtin_amdgcn_global_load_async_to_lds_b128) && \
    __has_builtin(__builtin_amdgcn_s_wait_asynccnt)
#define USE_ASYNC_LDS 1
#else
#define USE_ASYNC_LDS 0
#endif

#if USE_ASYNC_LDS
typedef v4i __attribute__((address_space(1))) g_v4i;   // global (AS1) v4i
typedef v4i __attribute__((address_space(3))) l_v4i;   // LDS    (AS3) v4i
__device__ __forceinline__ void async_copy16(const __bf16* g, __bf16* l) {
    __builtin_amdgcn_global_load_async_to_lds_b128((g_v4i*)g, (l_v4i*)l, 0, 0);
}
#endif

// ---------------------------------------------------------------------------
// Stage 1: fp32 -> bf16 conversion (4 elements / thread, 16B loads, 8B stores)
// ---------------------------------------------------------------------------
__global__ __launch_bounds__(256) void f32_to_bf16_kernel(
    const float* __restrict__ in, __bf16* __restrict__ out, int n)
{
    int i = (blockIdx.x * 256 + threadIdx.x) * 4;
    if (i >= n) return;
    float4 v = *(const float4*)(in + i);
    bf16x4 o = { (__bf16)v.x, (__bf16)v.y, (__bf16)v.z, (__bf16)v.w };
    *(bf16x4*)(out + i) = o;
}

// ---------------------------------------------------------------------------
// Stage 2: C[M,N] = A[M,K] * W[N,K]^T   (bf16 in, f32 out)
//   Block tile 128x128, BK=32, 256 threads = 8 waves (wave32).
//   Wave (wm,wn): wm in 0..3 covers 32 rows, wn in 0..1 covers 64 cols
//   -> 2x4 grid of 16x16 WMMA tiles per wave.
//   LDS double-buffered; filled with GLOBAL_LOAD_ASYNC_TO_LDS_B128 when
//   available, synchronized via s_wait_asynccnt.
// ---------------------------------------------------------------------------
#define BM 128
#define BN 128
#define BK 32
#define LSTRIDE 40   // 32 + 8 pad bf16 elems; 20-dword row stride -> no bank conflicts

__global__ __launch_bounds__(256) void gemm_bf16_nt_kernel(
    const __bf16* __restrict__ A, const __bf16* __restrict__ W,
    float* __restrict__ C, int M, int N, int K)
{
    __shared__ __bf16 lA[2][BM * LSTRIDE];
    __shared__ __bf16 lB[2][BN * LSTRIDE];

    const int tid  = threadIdx.x;
    const int lane = tid & 31;
    const int wave = tid >> 5;
    const int wm   = wave & 3;   // wave row  (M)
    const int wn   = wave >> 2;  // wave col  (N)
    const int half = lane >> 4;  // 0: lanes 0-15, 1: lanes 16-31
    const int lr   = lane & 15;

    const int blockN = blockIdx.x * BN;
    const int blockM = blockIdx.y * BM;

    v8f acc[2][4];
#pragma unroll
    for (int i = 0; i < 2; ++i)
#pragma unroll
        for (int j = 0; j < 4; ++j)
#pragma unroll
            for (int r = 0; r < 8; ++r) acc[i][j][r] = 0.0f;

    // Cooperative tile load mapping: 512 chunks of 8 bf16 (16B) per operand,
    // 2 chunks per thread per operand.
    const int row0 = tid >> 2;
    const int kc0  = tid & 3;
    const int row1 = (tid + 256) >> 2;
    const int kc1  = (tid + 256) & 3;

    const __bf16* pA0 = A + (size_t)(blockM + row0) * K + kc0 * 8;
    const __bf16* pA1 = A + (size_t)(blockM + row1) * K + kc1 * 8;
    const __bf16* pB0 = W + (size_t)(blockN + row0) * K + kc0 * 8;
    const __bf16* pB1 = W + (size_t)(blockN + row1) * K + kc1 * 8;
    const int dA0 = row0 * LSTRIDE + kc0 * 8;
    const int dA1 = row1 * LSTRIDE + kc1 * 8;
    const int dB0 = row0 * LSTRIDE + kc0 * 8;
    const int dB1 = row1 * LSTRIDE + kc1 * 8;

    // Per-K-step WMMA work on LDS buffer `buf`.
    auto compute_step = [&](int buf) {
        // A fragments (16x32 bf16, ISA layout):
        //   lanes 0-15 : K[0:8) in u[0], K[16:24) in u[1]
        //   lanes 16-31: K[8:16) in u[0], K[24:32) in u[1]
        FragU af[2];
#pragma unroll
        for (int i = 0; i < 2; ++i) {
            int r    = wm * 32 + i * 16 + lr;
            int base = r * LSTRIDE + half * 8;
            af[i].u[0] = *(const uint4*)&lA[buf][base];
            af[i].u[1] = *(const uint4*)&lA[buf][base + 16];
        }
        // B fragments (32x16 bf16, ISA layout):
        //   lanes 0-15 : K[0:8) in u[0], K[16:24) in u[1]
        //   lanes 16-31: K[8:16) in u[0], K[24:32) in u[1]
        FragU bfr[4];
#pragma unroll
        for (int j = 0; j < 4; ++j) {
            int r    = wn * 64 + j * 16 + lr;
            int base = r * LSTRIDE + half * 8;
            bfr[j].u[0] = *(const uint4*)&lB[buf][base];
            bfr[j].u[1] = *(const uint4*)&lB[buf][base + 16];
        }
#pragma unroll
        for (int i = 0; i < 2; ++i)
#pragma unroll
            for (int j = 0; j < 4; ++j)
                acc[i][j] = __builtin_amdgcn_wmma_f32_16x16x32_bf16(
                    /*neg_a=*/false, af[i].v, /*neg_b=*/false, bfr[j].v,
                    /*c_mod=*/(short)0, acc[i][j],
                    /*reuse_a=*/false, /*reuse_b=*/false);
    };

#if USE_ASYNC_LDS
    // ---- double-buffered async pipeline ----
    auto issue_copies = [&](int buf, int k0) {
        async_copy16(pA0 + k0, &lA[buf][dA0]);
        async_copy16(pA1 + k0, &lA[buf][dA1]);
        async_copy16(pB0 + k0, &lB[buf][dB0]);
        async_copy16(pB1 + k0, &lB[buf][dB1]);
    };

    const int nsteps = K / BK;
    issue_copies(0, 0);
    for (int s = 0; s < nsteps - 1; ++s) {
        const int cur = s & 1;
        issue_copies(cur ^ 1, (s + 1) * BK);          // prefetch next buffer
        __builtin_amdgcn_s_wait_asynccnt(4);          // current buffer landed
        __syncthreads();                              // visible to all waves
        compute_step(cur);
        __syncthreads();                              // all waves done reading
    }
    {
        const int cur = (nsteps - 1) & 1;
        __builtin_amdgcn_s_wait_asynccnt(0);
        __syncthreads();
        compute_step(cur);
    }
#else
    // ---- fallback: sync staging through VGPRs ----
    for (int k0 = 0, buf = 0; k0 < K; k0 += BK, buf ^= 1) {
        uint4 a0 = *(const uint4*)(pA0 + k0);
        uint4 a1 = *(const uint4*)(pA1 + k0);
        uint4 b0 = *(const uint4*)(pB0 + k0);
        uint4 b1 = *(const uint4*)(pB1 + k0);
        *(uint4*)&lA[buf][dA0] = a0;
        *(uint4*)&lA[buf][dA1] = a1;
        *(uint4*)&lB[buf][dB0] = b0;
        *(uint4*)&lB[buf][dB1] = b1;
        __syncthreads();
        compute_step(buf);
        __syncthreads();
    }
#endif

    // Epilogue: C/D layout — VGPR r: lanes 0-15 -> M = base+r, lanes 16-31 -> M = base+8+r
#pragma unroll
    for (int i = 0; i < 2; ++i) {
#pragma unroll
        for (int j = 0; j < 4; ++j) {
            int m0 = blockM + wm * 32 + i * 16 + half * 8;
            int n  = blockN + wn * 64 + j * 16 + lr;
            float* cp = C + (size_t)m0 * N + n;
#pragma unroll
            for (int r = 0; r < 8; ++r) cp[(size_t)r * N] = acc[i][j][r];
        }
    }
}

// ---------------------------------------------------------------------------
// Stage 3: per-row LayerNorm(igates) + LayerNorm(hgates) + LSTM elementwise
//          + LayerNorm(cy). One block (256 threads) per batch row.
// ---------------------------------------------------------------------------
__device__ __forceinline__ float sigmoid_f(float x) { return 1.0f / (1.0f + __expf(-x)); }

__global__ __launch_bounds__(256) void ln_lstm_kernel(
    const float* __restrict__ Ci, const float* __restrict__ Ch,
    const float* __restrict__ cx,
    const float* __restrict__ gi, const float* __restrict__ bi,
    const float* __restrict__ gh, const float* __restrict__ bh,
    const float* __restrict__ gc, const float* __restrict__ bc,
    float* __restrict__ out, int H4, int H)
{
    const int b   = blockIdx.x;
    const int tid = threadIdx.x;
    __shared__ float4 sred[8];
    __shared__ float4 sres;

    const float* ci = Ci + (size_t)b * H4;
    const float* ch = Ch + (size_t)b * H4;

    // ---- stats over the two 4096-wide gate rows ----
    float4 s = {0.f, 0.f, 0.f, 0.f};
    for (int c = tid; c < H4; c += 256) {
        float vi = ci[c], vh = ch[c];
        s.x += vi; s.y += vi * vi; s.z += vh; s.w += vh * vh;
    }
#pragma unroll
    for (int off = 16; off > 0; off >>= 1) {
        s.x += __shfl_down(s.x, off, 32);
        s.y += __shfl_down(s.y, off, 32);
        s.z += __shfl_down(s.z, off, 32);
        s.w += __shfl_down(s.w, off, 32);
    }
    if ((tid & 31) == 0) sred[tid >> 5] = s;
    __syncthreads();
    if (tid == 0) {
        float4 t = sred[0];
        for (int w = 1; w < 8; ++w) {
            t.x += sred[w].x; t.y += sred[w].y; t.z += sred[w].z; t.w += sred[w].w;
        }
        sres = t;
    }
    __syncthreads();
    const float invn = 1.0f / (float)H4;
    const float mu_i = sres.x * invn;
    const float rs_i = rsqrtf(sres.y * invn - mu_i * mu_i + 1e-5f);
    const float mu_h = sres.z * invn;
    const float rs_h = rsqrtf(sres.w * invn - mu_h * mu_h + 1e-5f);

    // ---- gates + pre-LN cell value ----
    float tval[4], oval[4];
    float scx = 0.f, scy = 0.f;
#pragma unroll
    for (int k = 0; k < 4; ++k) {
        int h  = tid + k * 256;
        int c1 = H + h, c2 = 2 * H + h, c3 = 3 * H + h;
        float g0 = (ci[h]  - mu_i) * rs_i * gi[h]  + bi[h]  + (ch[h]  - mu_h) * rs_h * gh[h]  + bh[h];
        float g1 = (ci[c1] - mu_i) * rs_i * gi[c1] + bi[c1] + (ch[c1] - mu_h) * rs_h * gh[c1] + bh[c1];
        float g2 = (ci[c2] - mu_i) * rs_i * gi[c2] + bi[c2] + (ch[c2] - mu_h) * rs_h * gh[c2] + bh[c2];
        float g3 = (ci[c3] - mu_i) * rs_i * gi[c3] + bi[c3] + (ch[c3] - mu_h) * rs_h * gh[c3] + bh[c3];
        float ig = sigmoid_f(g0), fg = sigmoid_f(g1);
        float gg = tanhf(g2),     og = sigmoid_f(g3);
        float t = fg * cx[(size_t)b * H + h] + ig * gg;
        tval[k] = t; oval[k] = og;
        scx += t; scy += t * t;
    }
#pragma unroll
    for (int off = 16; off > 0; off >>= 1) {
        scx += __shfl_down(scx, off, 32);
        scy += __shfl_down(scy, off, 32);
    }
    __syncthreads();   // everyone finished reading sres/sred above
    if ((tid & 31) == 0) { sred[tid >> 5].x = scx; sred[tid >> 5].y = scy; }
    __syncthreads();
    if (tid == 0) {
        float sx = 0.f, sy = 0.f;
        for (int w = 0; w < 8; ++w) { sx += sred[w].x; sy += sred[w].y; }
        sres.x = sx; sres.y = sy;
    }
    __syncthreads();
    const float invH = 1.0f / (float)H;
    const float mu_c = sres.x * invH;
    const float rs_c = rsqrtf(sres.y * invH - mu_c * mu_c + 1e-5f);

    const size_t BH = (size_t)gridDim.x * (size_t)H;
#pragma unroll
    for (int k = 0; k < 4; ++k) {
        int h = tid + k * 256;
        float cy = (tval[k] - mu_c) * rs_c * gc[h] + bc[h];
        float hy = oval[k] * tanhf(cy);
        size_t idx = (size_t)b * H + h;
        out[idx]          = hy;
        out[BH + idx]     = hy;
        out[2 * BH + idx] = cy;
    }
}

// ---------------------------------------------------------------------------
// Launch
// ---------------------------------------------------------------------------
extern "C" void kernel_launch(void* const* d_in, const int* in_sizes, int n_in,
                              void* d_out, int out_size, void* d_ws, size_t ws_size,
                              hipStream_t stream)
{
    (void)in_sizes; (void)n_in; (void)out_size; (void)ws_size;

    const int Bsz = 4096, Idim = 1024, Hdim = 1024, H4 = 4096;
    const int NELEM = Bsz * Idim;          // 4M, same for hx / w_ih / w_hh

    const float* input = (const float*)d_in[0];
    const float* hx    = (const float*)d_in[1];
    const float* cx    = (const float*)d_in[2];
    const float* wih   = (const float*)d_in[3];
    const float* whh   = (const float*)d_in[4];
    const float* gi    = (const float*)d_in[5];
    const float* bi    = (const float*)d_in[6];
    const float* gh    = (const float*)d_in[7];
    const float* bh    = (const float*)d_in[8];
    const float* gc    = (const float*)d_in[9];
    const float* bc    = (const float*)d_in[10];
    float* out = (float*)d_out;

    // Workspace layout (160 MB total):
    //   [0,8MB)   bf16 input     [8,16MB)  bf16 hx
    //   [16,24MB) bf16 W_ih      [24,32MB) bf16 W_hh
    //   [32,96MB) Ci f32 4096x4096   [96,160MB) Ch f32 4096x4096
    char* ws = (char*)d_ws;
    const size_t MB = 1ull << 20;
    __bf16* bIn  = (__bf16*)(ws + 0 * MB);
    __bf16* bHx  = (__bf16*)(ws + 8 * MB);
    __bf16* bWih = (__bf16*)(ws + 16 * MB);
    __bf16* bWhh = (__bf16*)(ws + 24 * MB);
    float*  Ci   = (float*)(ws + 32 * MB);
    float*  Ch   = (float*)(ws + 96 * MB);

    const int cvtBlocks = NELEM / 4 / 256;   // 4096
    f32_to_bf16_kernel<<<cvtBlocks, 256, 0, stream>>>(input, bIn,  NELEM);
    f32_to_bf16_kernel<<<cvtBlocks, 256, 0, stream>>>(hx,    bHx,  NELEM);
    f32_to_bf16_kernel<<<cvtBlocks, 256, 0, stream>>>(wih,   bWih, NELEM);
    f32_to_bf16_kernel<<<cvtBlocks, 256, 0, stream>>>(whh,   bWhh, NELEM);

    dim3 gg(H4 / BN, Bsz / BM);   // 32 x 32 blocks
    gemm_bf16_nt_kernel<<<gg, 256, 0, stream>>>(bIn, bWih, Ci, Bsz, H4, Idim);
    gemm_bf16_nt_kernel<<<gg, 256, 0, stream>>>(bHx, bWhh, Ch, Bsz, H4, Hdim);

    ln_lstm_kernel<<<Bsz, 256, 0, stream>>>(Ci, Ch, cx, gi, bi, gh, bh, gc, bc,
                                            out, H4, Hdim);
}
